// PQCodebookModel_59854664237415
// MI455X (gfx1250) — matrix operand order repaired and checked by reading, not compiled, not run on hardware
//
#include <hip/hip_runtime.h>
#include <hip/hip_bf16.h>

// ---- problem constants (from reference) ------------------------------------
#define NTOK   16384      // BATCH*SEQ
#define NSUB   16         // M
#define CBSZ   1024       // K
#define SDIM   64         // d
#define EMB    1024       // NSUB*SDIM

typedef __bf16 bf16_t;
typedef __attribute__((ext_vector_type(16))) __bf16 v16bf;
typedef __attribute__((ext_vector_type(8)))  float  v8f;

#define WMMA_BF16(A, B, C) \
  __builtin_amdgcn_wmma_f32_16x16x32_bf16(false, (A), false, (B), (short)0, (C), false, false)

// ----------------------------------------------------------------------------
// Kernel 1: repack codebooks into per-tile-contiguous WMMA B-fragment blocks,
// split bf16 hi/lo.  Per (subspace m, 16-codeword tile t): 4 fragments of
// 32 lanes x 16 halves, order f = {c0.hi, c0.lo, c1.hi, c1.lo} (c = 32-dim
// chunk) -> 2048 halves = 4KB, contiguous, so one tile = one async-copy block.
// B fragment layout used here (contiguous k, same order as A; 32x16, 8 VGPRs):
//   lanes 0-15  : column n = lane,    e=0..15 -> K(dim) = c*32 + e
//   lanes 16-31 : column n = lane-16, e=0..15 -> K(dim) = c*32 + 16 + e
// Flat slot = (((m*64 + t)*4 + f)*32 + lane)*16 + e ; total = 1<<21 halves.
// ----------------------------------------------------------------------------
__global__ __launch_bounds__(256) void pq_pack(const float* __restrict__ cb,
                                               bf16_t* __restrict__ pk) {
  unsigned gid = blockIdx.x * 256u + threadIdx.x;      // < 1<<21
  int e    =  gid        & 15;
  int lane = (gid >> 4)  & 31;
  int f    = (gid >> 9)  & 3;
  int t    = (gid >> 11) & 63;
  int m    =  gid >> 17;
  int c    = f >> 1;
  int isLo = f & 1;
  int k    = t * 16 + (lane & 15);
  int dim  = c * 32 + ((lane >> 4) * 16) + e;
  float x  = cb[((size_t)m * CBSZ + k) * SDIM + dim];
  bf16_t h = (bf16_t)x;
  pk[gid]  = isLo ? (bf16_t)(x - (float)h) : h;
}

// ---- Kernel 2: codeword squared norms, c2[m*CBSZ + k] ----------------------
__global__ __launch_bounds__(256) void pq_c2(const float* __restrict__ cb,
                                             float* __restrict__ c2) {
  unsigned gid = blockIdx.x * 256u + threadIdx.x;      // < NSUB*CBSZ
  const float* p = cb + (size_t)gid * SDIM;
  float s = 0.f;
#pragma unroll
  for (int j = 0; j < SDIM; ++j) s = fmaf(p[j], p[j], s);
  c2[gid] = s;
}

// ----------------------------------------------------------------------------
// Kernel 3: fused scores + argmin + gather.
// Block = 8 waves, one subspace m, 256 tokens; each wave owns 32 tokens as two
// 16-row WMMA tiles (12 WMMAs per B tile, 4 independent accumulator chains).
// B tiles (4KB) are staged global->LDS with async copies, double-buffered:
// all 8 waves share each staged tile (8x less L2 operand traffic).
// A fragment layout for 16-bit A (16x32):
//   lanes 0-15 : row M = lane,    e0..15 -> K=c*32+0..15 (contiguous, same k order as B)
//   lanes16-31 : row M = lane-16, e0..15 -> K=c*32+16..31
// ----------------------------------------------------------------------------
__global__ __launch_bounds__(256) void pq_main(
    const float*  __restrict__ emb,
    const float*  __restrict__ cbf32,
    const bf16_t* __restrict__ cb_pk,
    const float*  __restrict__ c2,
    float*        __restrict__ outq,
    long long*    __restrict__ outidx) {
  __shared__ __align__(16) bf16_t sB[2][2048];          // 2 x 4KB double buffer

  const int tid     = threadIdx.x;
  const int lane    = tid & 31;
  const int w       = tid >> 5;
  const int m       = blockIdx.x & (NSUB - 1);
  const int sb      = blockIdx.x >> 4;                  // 0..63
  const int wbase   = sb * 256 + w * 32;                // first of 32 tokens
  const int lanecol = lane & 15;
  const int lanehi  = lane >> 4;

  // ---- build A fragments (split bf16): 2 token tiles x 2 k-chunks
  v16bf Ah[2][2], Al[2][2];
#pragma unroll
  for (int tile = 0; tile < 2; ++tile) {
    const int rowA = wbase + tile * 16 + lanecol;
    const float* xrow = emb + (size_t)rowA * EMB + m * SDIM;
#pragma unroll
    for (int c = 0; c < 2; ++c) {
      int g0 = c * 32 + lanehi * 16;
#pragma unroll
      for (int e = 0; e < 8; ++e) {
        float x0 = xrow[g0 + e];
        bf16_t h0 = (bf16_t)x0;
        Ah[tile][c][e] = h0;
        Al[tile][c][e] = (bf16_t)(x0 - (float)h0);
        float x1 = xrow[g0 + 8 + e];
        bf16_t h1 = (bf16_t)x1;
        Ah[tile][c][8 + e] = h1;
        Al[tile][c][8 + e] = (bf16_t)(x1 - (float)h1);
      }
    }
  }

  float minv[2][8];
  int   mini[2][8];
#pragma unroll
  for (int tile = 0; tile < 2; ++tile)
#pragma unroll
    for (int r = 0; r < 8; ++r) { minv[tile][r] = 3.402823466e38f; mini[tile][r] = 0; }

  // per-subspace packed codebook: 64 tiles x 2048 halves (4KB each)
  const bf16_t* gB = cb_pk + (size_t)m * 64 * 2048;

  // each of the 256 threads async-copies 16 bytes of the 4KB tile
  auto issue = [&](int t, int buf) {
    const bf16_t* src = gB + (size_t)t * 2048 + tid * 8;
    unsigned ldsa = (unsigned)(size_t)(const void*)(&sB[buf][tid * 8]);
    asm volatile("global_load_async_to_lds_b128 %0, %1, off"
                 :: "v"(ldsa), "v"(src) : "memory");
  };

  issue(0, 0);
  for (int t = 0; t < 64; ++t) {
    if (t < 63) {
      issue(t + 1, (t + 1) & 1);
      // per-wave ASYNCcnt completes in order: <=1 outstanding => tile t landed
      asm volatile("s_wait_asynccnt 0x1" ::: "memory");
    } else {
      asm volatile("s_wait_asynccnt 0x0" ::: "memory");
    }
    __syncthreads();   // all waves' slices of tile t are in LDS

    const bf16_t* bb = &sB[t & 1][0];
    v16bf bhi0 = *(const v16bf*)(bb +    0 + lane * 16);
    v16bf blo0 = *(const v16bf*)(bb +  512 + lane * 16);
    v16bf bhi1 = *(const v16bf*)(bb + 1024 + lane * 16);
    v16bf blo1 = *(const v16bf*)(bb + 1536 + lane * 16);

    // 4 independent accumulator chains (2 token tiles x {hi.hi, corrections})
    v8f a00 = {}, a01 = {}, a10 = {}, a11 = {};
    a00 = WMMA_BF16(Ah[0][0], bhi0, a00);
    a01 = WMMA_BF16(Al[0][0], bhi0, a01);
    a10 = WMMA_BF16(Ah[1][0], bhi0, a10);
    a11 = WMMA_BF16(Al[1][0], bhi0, a11);
    a00 = WMMA_BF16(Ah[0][1], bhi1, a00);
    a01 = WMMA_BF16(Al[0][1], bhi1, a01);
    a10 = WMMA_BF16(Ah[1][1], bhi1, a10);
    a11 = WMMA_BF16(Al[1][1], bhi1, a11);
    a00 = WMMA_BF16(Ah[0][0], blo0, a00);
    a01 = WMMA_BF16(Ah[0][1], blo1, a01);
    a10 = WMMA_BF16(Ah[1][0], blo0, a10);
    a11 = WMMA_BF16(Ah[1][1], blo1, a11);

    const int   kcol = t * 16 + lanecol;
    const float c2v  = c2[m * CBSZ + kcol];
#pragma unroll
    for (int r = 0; r < 8; ++r) {
      float d0 = fmaf(-2.0f, a00[r] + a01[r], c2v);   // ||x||^2 constant: drop
      float d1 = fmaf(-2.0f, a10[r] + a11[r], c2v);
      if (d0 < minv[0][r]) { minv[0][r] = d0; mini[0][r] = kcol; }
      if (d1 < minv[1][r]) { minv[1][r] = d1; mini[1][r] = kcol; }
    }
    __syncthreads();   // everyone done reading buf[t&1] before it is re-filled
  }

  // ---- cross-lane argmin within each 16-lane half (columns), index tie-break
#pragma unroll
  for (int tile = 0; tile < 2; ++tile)
#pragma unroll
    for (int r = 0; r < 8; ++r)
#pragma unroll
      for (int mask = 1; mask < 16; mask <<= 1) {
        float ov = __shfl_xor(minv[tile][r], mask, 32);
        int   oi = __shfl_xor(mini[tile][r], mask, 32);
        if (ov < minv[tile][r] || (ov == minv[tile][r] && oi < mini[tile][r])) {
          minv[tile][r] = ov; mini[tile][r] = oi;
        }
      }
  // lanes 0-15 agree on rows 0..7 of each tile; lanes 16-31 on rows 8..15

  // ---- write int64 indices (lane 0 -> rows 0..7, lane 16 -> rows 8..15)
  if (lanecol == 0) {
    const int rbase = lanehi * 8;
#pragma unroll
    for (int tile = 0; tile < 2; ++tile)
#pragma unroll
      for (int r = 0; r < 8; ++r) {
        int row = wbase + tile * 16 + rbase + r;
        outidx[(size_t)row * NSUB + m] = (long long)mini[tile][r];
      }
  }

  // ---- broadcast winners, gather exact f32 codewords (2 floats/lane/row)
  int best[2][16];
#pragma unroll
  for (int tile = 0; tile < 2; ++tile)
#pragma unroll
    for (int r = 0; r < 8; ++r) {
      best[tile][r]     = __shfl(mini[tile][r], 0, 32);
      best[tile][8 + r] = __shfl(mini[tile][r], 16, 32);
    }
#pragma unroll
  for (int tile = 0; tile < 2; ++tile)
#pragma unroll
    for (int rr = 0; rr < 16; ++rr) {
      int row = wbase + tile * 16 + rr;
      const float* src = cbf32 + ((size_t)m * CBSZ + best[tile][rr]) * SDIM;
      float2 v = *(const float2*)(src + lane * 2);
      *(float2*)(outq + (size_t)row * EMB + m * SDIM + lane * 2) = v;
    }
}

// ----------------------------------------------------------------------------
extern "C" void kernel_launch(void* const* d_in, const int* in_sizes, int n_in,
                              void* d_out, int out_size, void* d_ws, size_t ws_size,
                              hipStream_t stream) {
  (void)in_sizes; (void)n_in; (void)out_size; (void)ws_size;
  const float* emb = (const float*)d_in[0];   // [NTOK, EMB] f32
  const float* cb  = (const float*)d_in[1];   // [NSUB, CBSZ, SDIM] f32

  float*     outq   = (float*)d_out;                                      // [NTOK, EMB] f32
  long long* outidx = (long long*)((char*)d_out + (size_t)NTOK * EMB * 4); // [NTOK, NSUB] i64

  // workspace: cb_pk (4MB, hi/lo interleaved per tile) | c2 (64KB)
  const size_t NSLOT = (size_t)NSUB * CBSZ * SDIM * 2;  // 1<<21 halves
  bf16_t* cb_pk = (bf16_t*)d_ws;
  float*  c2    = (float*)((char*)d_ws + 2 * NSLOT);

  pq_pack<<<(unsigned)(NSLOT / 256), 256, 0, stream>>>(cb, cb_pk);
  pq_c2<<<(NSUB * CBSZ) / 256, 256, 0, stream>>>(cb, c2);
  // block = (subspace, 256-token super-tile); 8 waves x 32 tokens each
  pq_main<<<NSUB * (NTOK / 256), 256, 0, stream>>>(emb, cb, cb_pk, c2,
                                                   outq, outidx);
}
